// RoIHead_61280593379394
// MI455X (gfx1250) — matrix operand order repaired and checked by reading, not compiled, not run on hardware
//
#include <hip/hip_runtime.h>

typedef __bf16 v16bf __attribute__((ext_vector_type(16)));
typedef float  v8f   __attribute__((ext_vector_type(8)));

#define ROISIZE 7
#define HF      50
#define WF      50
#define RNUM    300
#define MPAD    304          // 19 * 16
#define MT      19           // M tiles (16 rows each)
#define MTPW    10           // M tiles per wave (2 m-groups: 10 + 9)
#define DIN     25088        // 512 * 49
#define DH      4096
#define NCLS    81
#define NLOC    324
#define KS      8            // K-split factor for the big GEMMs (occupancy)

// ---- fp32 -> bf16 helpers ----
// RNE (used for activation stores, cold path)
static __device__ __forceinline__ unsigned int f2bf_u(float x) {
  unsigned int u = __builtin_bit_cast(unsigned int, x);
  u += 0x7FFFu + ((u >> 16) & 1u);
  return u >> 16;
}
// fast pair-pack (round-half-up + v_perm_b32), hot path
static __device__ __forceinline__ unsigned int pk_bf16(float lo, float hi) {
  unsigned int a = __builtin_bit_cast(unsigned int, lo) + 0x8000u;
  unsigned int b = __builtin_bit_cast(unsigned int, hi) + 0x8000u;
  // D bytes [3,2] = b[3,2] (sel 7,6) ; D bytes [1,0] = a[3,2] (sel 3,2)
  return __builtin_amdgcn_perm(b, a, 0x07060302u);
}

// WMMA A/B fragment (8 dwords = 16 bf16 per lane)
union AFrag { unsigned int u[8]; v16bf v; };

// A fragment, contiguous K order: dword v -> K pair (2v+16h)
//  => dword group {8h..8h+7}.
static __device__ __forceinline__ void load_afrag(AFrag& f, const unsigned int* At) {
#pragma unroll
  for (int v = 0; v < 4; ++v) f.u[v] = At[v];
#pragma unroll
  for (int v = 0; v < 4; ++v) f.u[4 + v] = At[4 + v];
}

// ---------------------------------------------------------------------------
// Kernel 1: RoI max pool -> bf16 activation matrix A [MPAD x DIN], row-major.
// Rows >= RNUM are zero-filled padding. Bin bounds match the reference:
// b = round(box/16); start = c1 + i*len/7; end = c1 + ceil((i+1)*len/7),
// clipped to [0,50]; empty bin -> -inf.
// ---------------------------------------------------------------------------
__global__ void roi_pool_bf16_kernel(const float* __restrict__ feat,
                                     const float* __restrict__ boxes,
                                     unsigned short* __restrict__ A) {
  const int r = blockIdx.x;
  __shared__ int sh[4][ROISIZE];   // 0=hs 1=he 2=ws 3=we
  if (r < RNUM && threadIdx.x < 4 * ROISIZE) {
    const int d = threadIdx.x / ROISIZE;
    const int i = threadIdx.x % ROISIZE;
    const float a1 = boxes[r * 4 + ((d < 2) ? 0 : 1)];
    const float a2 = boxes[r * 4 + ((d < 2) ? 2 : 3)];
    const int c1 = (int)rintf(a1 * 0.0625f);
    const int c2 = (int)rintf(a2 * 0.0625f);
    int len = c2 - c1 + 1;
    if (len < 1) len = 1;
    int v = ((d & 1) == 0) ? (c1 + (i * len) / ROISIZE)
                           : (c1 + ((i + 1) * len + ROISIZE - 1) / ROISIZE);
    v = min(max(v, 0), HF);
    sh[d][i] = v;
  }
  __syncthreads();

  unsigned int* __restrict__ Arow = (unsigned int*)A + (size_t)r * (DIN / 2);
  for (int p = threadIdx.x; p < DIN / 2; p += blockDim.x) {
    unsigned int pack = 0u;
    if (r < RNUM) {
      float vv[2];
#pragma unroll
      for (int e = 0; e < 2; ++e) {
        const int col = 2 * p + e;
        const int c   = col / 49;
        const int rem = col - c * 49;
        const int i   = rem / 7;
        const int j   = rem - i * 7;
        const int y0 = sh[0][i], y1 = sh[1][i];
        const int x0 = sh[2][j], x1 = sh[3][j];
        const float* __restrict__ base = feat + (size_t)c * (HF * WF);
        float m = -INFINITY;
        for (int y = y0; y < y1; ++y)
          for (int x = x0; x < x1; ++x)
            m = fmaxf(m, base[y * WF + x]);
        vv[e] = m;
      }
      pack = (f2bf_u(vv[0]) & 0xFFFFu) | (f2bf_u(vv[1]) << 16);
    }
    Arow[p] = pack;
  }
}

// ---------------------------------------------------------------------------
// Kernel 2: big GEMM partials  P[ks] = A * B  for one K-chunk.
//   A: bf16 [MPAD x K] row-major; B: fp32 [K x N] row-major; N % 128 == 0,
//   K % (32*KS) == 0. Block = 512 threads (16 waves) = 8 N-tiles x 2 M-groups;
//   each wave accumulates 9-10 M-tiles, so every weight element is loaded
//   from global exactly once per K-chunk, converted to bf16 once, and
//   broadcast to all 16 waves via LDS (padded -> conflict-free ds_load_b128).
//   Software pipelining: the next B chunk's global loads are issued before
//   the barrier/consume phase, and A fragments are double-buffered so a
//   global_load_b128 pair is always in flight under each WMMA.
// ---------------------------------------------------------------------------
__global__ __launch_bounds__(512)
void gemm_big_wmma(const unsigned short* __restrict__ A,
                   const float* __restrict__ B,
                   float* __restrict__ P,
                   int N, int K) {
  __shared__ unsigned int ldsB[128 * 20];  // 128 cols x 16 dwords, pad to 20

  const int tid  = threadIdx.x;
  const int wave = tid >> 5;
  const int lane = tid & 31;
  const int l16  = lane & 15;
  const int h    = lane >> 4;
  const int ntl  = wave & 7;                       // n-tile within block
  const int mg   = wave >> 3;                      // m-group (0 or 1)
  const int mcnt = (mg == 0) ? MTPW : (MT - MTPW); // 10 or 9
  const int n0b  = blockIdx.x * 128;

  const int kchunk = K / KS;
  const int kstart = blockIdx.y * kchunk;
  const int kend   = kstart + kchunk;

  // B staging assignment: thread -> (local col, 4 consecutive k-pairs)
  const int sc  = tid & 127;
  const int kpb = (tid >> 7) << 2;                 // 0,4,8,12
  const float* __restrict__ Bcol = B + n0b + sc;

  // A base for this lane's first m-tile row
  const unsigned int* __restrict__ A0 =
      (const unsigned int*)(A + (size_t)((mg * MTPW) * 16 + l16) * K);
  const size_t tileStride = (size_t)8 * K;         // dwords per 16-row m-tile

  v8f zero = {};
  v8f acc[MTPW];
#pragma unroll
  for (int t = 0; t < MTPW; ++t) acc[t] = zero;

  const int ldsw = (ntl * 16 + l16) * 20 + 8 * h;  // fragment read base

  // prologue: load first B chunk into registers
  float s0[4], s1[4];
#pragma unroll
  for (int j = 0; j < 4; ++j) {
    const int k = kstart + 2 * (kpb + j);
    s0[j] = Bcol[(size_t)k * N];
    s1[j] = Bcol[(size_t)(k + 1) * N];
  }

  for (int kk = kstart; kk < kend; kk += 32) {
    __syncthreads();                       // previous consume done
    // pack current chunk (registers) -> LDS
#pragma unroll
    for (int j = 0; j < 4; ++j)
      ldsB[sc * 20 + kpb + j] = pk_bf16(s0[j], s1[j]);
    // issue next chunk's loads now: they fly during barrier + consume
    if (kk + 32 < kend) {
#pragma unroll
      for (int j = 0; j < 4; ++j) {
        const int k = kk + 32 + 2 * (kpb + j);
        s0[j] = Bcol[(size_t)k * N];
        s1[j] = Bcol[(size_t)(k + 1) * N];
      }
    }
    if (kk + 96 < kend)                    // L2 prefetch two steps ahead
      __builtin_prefetch(&Bcol[(size_t)(kk + 64 + 2 * kpb) * N], 0, 1);
    __syncthreads();                       // LDS stores visible

    // B fragment: lane = column, dword v holds K pair (kk + 2v + 16h)
    AFrag bfm;
#pragma unroll
    for (int v = 0; v < 8; ++v) bfm.u[v] = ldsB[ldsw + v];

    const unsigned int* Ak = A0 + (kk >> 1) + 8 * h;
    AFrag afr[2];
    load_afrag(afr[0], Ak);
#pragma unroll
    for (int t = 0; t < MTPW; ++t) {
      if (t < mcnt) {
        if (t + 1 < mcnt)
          load_afrag(afr[(t + 1) & 1], Ak + (size_t)(t + 1) * tileStride);
        acc[t] = __builtin_amdgcn_wmma_f32_16x16x32_bf16(
            false, afr[t & 1].v, false, bfm.v, (short)0, acc[t], false, false);
      }
    }
  }

  // write fp32 partials for this K-chunk
  const int nc = n0b + ntl * 16 + l16;
  float* __restrict__ Pk = P + (size_t)blockIdx.y * MPAD * N;
#pragma unroll
  for (int t = 0; t < MTPW; ++t) {
    if (t < mcnt) {
#pragma unroll
      for (int r = 0; r < 8; ++r) {
        const int mr = (mg * MTPW + t) * 16 + r + 8 * h;
        Pk[(size_t)mr * N + nc] = acc[t][r];
      }
    }
  }
}

// ---------------------------------------------------------------------------
// Kernel 3: deterministic K-split reduction + bias + ReLU -> bf16 activations.
// N must be a power of two (4096 here).
// ---------------------------------------------------------------------------
__global__ void reduce_bias_relu_bf16(const float* __restrict__ P,
                                      const float* __restrict__ bias,
                                      unsigned short* __restrict__ F,
                                      int N) {
  const size_t i = (size_t)blockIdx.x * blockDim.x + threadIdx.x;
  const size_t total = (size_t)MPAD * N;
  if (i >= total) return;
  const int n = (int)(i & (size_t)(N - 1));
  float s = 0.0f;
#pragma unroll
  for (int ks = 0; ks < KS; ++ks) s += P[(size_t)ks * total + i];
  s += bias[n];
  s = fmaxf(s, 0.0f);
  F[i] = (unsigned short)f2bf_u(s);
}

// ---------------------------------------------------------------------------
// Kernel 4: small-N head GEMM (score/loc): one 16x16 tile per wave.
//   D = A * B + bias (fp32 out). Edge tiles clamp loads, guard stores.
// ---------------------------------------------------------------------------
__global__ void gemm_head_wmma(const unsigned short* __restrict__ A,
                               const float* __restrict__ B,
                               const float* __restrict__ bias,
                               float* __restrict__ outF,
                               int M, int N, int K) {
  const int wave = threadIdx.x >> 5;
  const int lane = threadIdx.x & 31;
  const int nt = blockIdx.y * 8 + wave;
  const int n0 = nt * 16;
  if (n0 >= N) return;
  const int m0  = blockIdx.x * 16;
  const int l16 = lane & 15;
  const int h   = lane >> 4;

  const int m  = m0 + l16;
  const int nb = min(n0 + l16, N - 1);

  const unsigned int* __restrict__ Abase = (const unsigned int*)(A + (size_t)m * K);

  v8f acc = {};
  for (int kk = 0; kk < K; kk += 32) {
    AFrag af, bfm;
    load_afrag(af, Abase + (kk >> 1) + 8 * h);
#pragma unroll
    for (int v = 0; v < 8; ++v) {
      const int k = kk + 2 * v + 16 * h;
      const float f0 = B[(size_t)k * N + nb];
      const float f1 = B[(size_t)(k + 1) * N + nb];
      bfm.u[v] = pk_bf16(f0, f1);
    }
    acc = __builtin_amdgcn_wmma_f32_16x16x32_bf16(
        false, af.v, false, bfm.v, (short)0, acc, false, false);
  }

  const int nc = n0 + l16;
  const float bv = bias[nb];
#pragma unroll
  for (int r = 0; r < 8; ++r) {
    const int mr = m0 + r + 8 * h;
    if (mr < M && nc < N) outF[(size_t)mr * N + nc] = acc[r] + bv;
  }
}

// ---------------------------------------------------------------------------
// Launch: roi-pool -> fc6 (partials + reduce/relu) -> fc7 -> {score, loc}.
// Workspace: A1 bf16 15.25MB | F1 2.49MB | F2 2.49MB | P fp32 39.8MB.
// ---------------------------------------------------------------------------
extern "C" void kernel_launch(void* const* d_in, const int* in_sizes, int n_in,
                              void* d_out, int out_size, void* d_ws, size_t ws_size,
                              hipStream_t stream) {
  (void)in_sizes; (void)n_in; (void)out_size; (void)ws_size;

  const float* feat  = (const float*)d_in[0];
  const float* boxes = (const float*)d_in[1];
  const float* W1 = (const float*)d_in[2];
  const float* b1 = (const float*)d_in[3];
  const float* W2 = (const float*)d_in[4];
  const float* b2 = (const float*)d_in[5];
  const float* Wc = (const float*)d_in[6];
  const float* bc = (const float*)d_in[7];
  const float* Wl = (const float*)d_in[8];
  const float* bl = (const float*)d_in[9];
  float* out = (float*)d_out;

  char* ws = (char*)d_ws;
  size_t off = 0;
  unsigned short* A1 = (unsigned short*)(ws + off); off += (size_t)MPAD * DIN * 2;
  unsigned short* F1 = (unsigned short*)(ws + off); off += (size_t)MPAD * DH * 2;
  unsigned short* F2 = (unsigned short*)(ws + off); off += (size_t)MPAD * DH * 2;
  float*          P  = (float*)(ws + off);  // KS * MPAD * DH * 4

  roi_pool_bf16_kernel<<<dim3(MPAD), dim3(256), 0, stream>>>(feat, boxes, A1);

  // fc6: [304,25088] x [25088,4096] -> partials -> +b1, relu -> F1 (bf16)
  gemm_big_wmma<<<dim3(DH / 128, KS), dim3(512), 0, stream>>>(A1, W1, P, DH, DIN);
  reduce_bias_relu_bf16<<<dim3((MPAD * DH) / 256), dim3(256), 0, stream>>>(P, b1, F1, DH);

  // fc7: [304,4096] x [4096,4096] -> partials -> +b2, relu -> F2 (bf16)
  gemm_big_wmma<<<dim3(DH / 128, KS), dim3(512), 0, stream>>>(F1, W2, P, DH, DH);
  reduce_bias_relu_bf16<<<dim3((MPAD * DH) / 256), dim3(256), 0, stream>>>(P, b2, F2, DH);

  // heads -> d_out = concat(score [300x81], loc [300x324])
  gemm_head_wmma<<<dim3(MT, (NCLS + 127) / 128), dim3(256), 0, stream>>>(
      F2, Wc, bc, out, RNUM, NCLS, DH);
  gemm_head_wmma<<<dim3(MT, (NLOC + 127) / 128), dim3(256), 0, stream>>>(
      F2, Wl, bl, out + (size_t)RNUM * NCLS, RNUM, NLOC, DH);
}
